// CausalSelfAttention_17532056502780
// MI455X (gfx1250) — matrix operand order repaired and checked by reading, not compiled, not run on hardware
//
#include <hip/hip_runtime.h>
#include <stdint.h>

typedef __attribute__((ext_vector_type(16))) __bf16 v16bf;
typedef __attribute__((ext_vector_type(8)))  __bf16 v8bf;
typedef __attribute__((ext_vector_type(4)))  __bf16 v4bf;
typedef __attribute__((ext_vector_type(2)))  __bf16 bf16x2;
typedef __attribute__((ext_vector_type(8)))  float  v8f;

#define WMMA_BF16(A, B, C) \
  __builtin_amdgcn_wmma_f32_16x16x32_bf16(false, (A), false, (B), (short)0, (C), false, false)

static __device__ __forceinline__ __bf16 f2bf(float f) {
  unsigned u = __builtin_bit_cast(unsigned, f);
  unsigned r = u + 0x7FFFu + ((u >> 16) & 1u);
  unsigned short h = (unsigned short)(r >> 16);
  return __builtin_bit_cast(__bf16, h);
}

// CDNA5 async copy: global -> LDS, 16B per lane, tracked by ASYNCcnt.
static __device__ __forceinline__ void async_b128(void* lds_ptr, const void* gptr) {
  uint32_t l = (uint32_t)(uintptr_t)lds_ptr;   // low 32 bits = wave LDS offset
  uint64_t g = (uint64_t)(uintptr_t)gptr;
  asm volatile("global_load_async_to_lds_b128 %0, %1, off"
               :: "v"(l), "v"(g) : "memory");
}
static __device__ __forceinline__ void wait_async0() {
  asm volatile("s_wait_asynccnt 0x0" ::: "memory");
}

// A-fragment element offset pattern (contiguous k = 16*half + 2*v, matching B)
static __device__ __forceinline__ int a_k0(int v, int half) {
  return (half << 4) + (v << 1);
}

// ---------------------------------------------------------------------------
// Kernel 1: f32 -> bf16 (vectorized)
// ---------------------------------------------------------------------------
__global__ void k_f32_to_bf16(const float* __restrict__ src,
                              __bf16* __restrict__ dst, int n) {
  int i = blockIdx.x * blockDim.x + threadIdx.x;
  int stride = gridDim.x * blockDim.x;
  int n4 = n >> 2;
  for (; i < n4; i += stride) {
    float4 f = ((const float4*)src)[i];
    v4bf o;
    o[0] = f2bf(f.x); o[1] = f2bf(f.y); o[2] = f2bf(f.z); o[3] = f2bf(f.w);
    *(v4bf*)(dst + 4 * (size_t)i) = o;
  }
}

// ---------------------------------------------------------------------------
// Kernel 2: f32 -> bf16 + transpose (W [K][N] -> Wt [N][K]), 32x32 tiles
// ---------------------------------------------------------------------------
__global__ __launch_bounds__(256) void k_convt(const float* __restrict__ W,
                                               __bf16* __restrict__ Wt,
                                               int Kd, int Nd) {
  __shared__ __bf16 tile[32][34];
  const int n0 = blockIdx.x * 32;
  const int k0 = blockIdx.y * 32;
#pragma unroll
  for (int p = 0; p < 4; ++p) {
    int idx = threadIdx.x + p * 256;
    int kr = idx >> 5, nc = idx & 31;
    tile[kr][nc] = f2bf(W[(size_t)(k0 + kr) * Nd + n0 + nc]);
  }
  __syncthreads();
#pragma unroll
  for (int p = 0; p < 4; ++p) {
    int idx = threadIdx.x + p * 256;
    int nr = idx >> 5, kc = idx & 31;
    Wt[(size_t)(n0 + nr) * Kd + k0 + kc] = tile[kc][nr];
  }
}

// ---------------------------------------------------------------------------
// Kernel 3: bf16 GEMM  out = A[MxK] * Wt[NxK]^T + bias
//   128x128 block tile, 8 waves (2Mx4N 16x16 subtiles each), K-step 64:
//   16 WMMAs per wave per barrier. Double-buffered async-to-LDS staging.
//   mode 0: f32 row-major; 1: bf16 row-major; 2: bf16 V-transposed [B][H][64][T].
// ---------------------------------------------------------------------------
__global__ __launch_bounds__(256) void k_gemm_bf16(
    const __bf16* __restrict__ A,   // [M][K]
    const __bf16* __restrict__ Wt,  // [N][K]
    const float*  __restrict__ bias,// [N]
    float*        __restrict__ outF,
    __bf16*       __restrict__ outB,
    int M, int N, int K, int mode, int T, int H) {
  const int tid  = threadIdx.x;
  const int lane = tid & 31;
  const int wave = tid >> 5;
  const int half = lane >> 4;
  const int ln   = lane & 15;

  const int n0 = blockIdx.x * 128;
  const int m0 = blockIdx.y * 128;
  const int wm = wave & 3;        // M group (32 rows)
  const int wn = wave >> 2;       // N group (64 cols)

  __shared__ __align__(16) __bf16 lds_a[2][128][72];  // row stride 144B
  __shared__ __align__(16) __bf16 lds_w[2][128][72];

  // staging: 2 threads per row, each thread 4 b128 chunks per tensor
  const int sr = tid >> 1;            // 0..127
  const int sc = (tid & 1) * 8;       // 0 or 8
  const __bf16* gA = A  + (size_t)(m0 + sr) * K + sc;
  const __bf16* gW = Wt + (size_t)(n0 + sr) * K + sc;

  auto stage = [&](int buf, int k0) {
#pragma unroll
    for (int c = 0; c < 4; ++c) {
      async_b128(&lds_a[buf][sr][sc + 16 * c], gA + k0 + 16 * c);
      async_b128(&lds_w[buf][sr][sc + 16 * c], gW + k0 + 16 * c);
    }
  };

  v8f acc[2][4] = {{{}, {}, {}, {}}, {{}, {}, {}, {}}};

  stage(0, 0);
  for (int k0 = 0; k0 < K; k0 += 64) {
    const int buf = (k0 >> 6) & 1;
    wait_async0();
    __syncthreads();                       // buf ready everywhere; buf^1 free
    if (k0 + 64 < K) stage(buf ^ 1, k0 + 64);

    // A fragments: 2 M-subtiles x 2 k-steps
    v16bf af[2][2];
#pragma unroll
    for (int t = 0; t < 2; ++t) {
      const int row = wm * 32 + t * 16 + ln;
#pragma unroll
      for (int s = 0; s < 2; ++s) {
#pragma unroll
        for (int v = 0; v < 8; ++v) {
          bf16x2 d = *(const bf16x2*)&lds_a[buf][row][s * 32 + a_k0(v, half)];
          af[t][s][2 * v] = d.x; af[t][s][2 * v + 1] = d.y;
        }
      }
    }
#pragma unroll
    for (int j = 0; j < 4; ++j) {
      const int col = wn * 64 + j * 16 + ln;
      v16bf bfg[2];
#pragma unroll
      for (int s = 0; s < 2; ++s)
#pragma unroll
        for (int e = 0; e < 8; ++e) {
          bf16x2 d = *(const bf16x2*)&lds_w[buf][col][s * 32 + half * 16 + 2 * e];
          bfg[s][2 * e] = d.x; bfg[s][2 * e + 1] = d.y;
        }
      acc[0][j] = WMMA_BF16(af[0][0], bfg[0], acc[0][j]);
      acc[0][j] = WMMA_BF16(af[0][1], bfg[1], acc[0][j]);
      acc[1][j] = WMMA_BF16(af[1][0], bfg[0], acc[1][j]);
      acc[1][j] = WMMA_BF16(af[1][1], bfg[1], acc[1][j]);
    }
  }

  // epilogue
#pragma unroll
  for (int t = 0; t < 2; ++t) {
    const int gm0 = m0 + wm * 32 + t * 16 + (half << 3);  // first of 8 rows
    const int bidx = gm0 / T;
    const int tp   = gm0 - bidx * T;
#pragma unroll
    for (int j = 0; j < 4; ++j) {
      const int gn = n0 + wn * 64 + j * 16 + ln;
      const float bb = bias[gn];
      if (mode == 0) {
#pragma unroll
        for (int r = 0; r < 8; ++r)
          outF[(size_t)(gm0 + r) * N + gn] = acc[t][j][r] + bb;
      } else if (mode == 1) {
#pragma unroll
        for (int r = 0; r < 8; ++r)
          outB[(size_t)(gm0 + r) * N + gn] = f2bf(acc[t][j][r] + bb);
      } else {
        v8bf pk;
#pragma unroll
        for (int r = 0; r < 8; ++r) pk[r] = f2bf(acc[t][j][r] + bb);
        size_t idx = ((size_t)(bidx * H + (gn >> 6)) * 64 + (gn & 63)) * T + tp;
        *(v8bf*)(outB + idx) = pk;     // 16B store along T
      }
    }
  }
}

// ---------------------------------------------------------------------------
// Kernel 4: causal flash attention, D=64. 8 waves x 16 q-rows, 32-key chunks,
// double-buffered async K / Vt tiles, f32 online softmax, bf16 WMMA.
// ---------------------------------------------------------------------------
__global__ __launch_bounds__(256) void k_attn(
    const __bf16* __restrict__ Q,   // [B*T][C]
    const __bf16* __restrict__ Km,  // [B*T][C]
    const __bf16* __restrict__ Vt,  // [B*H][64][T]
    __bf16* __restrict__ Y,         // [B*T][C]
    int T, int C, int H) {
  const int tid  = threadIdx.x;
  const int lane = tid & 31;
  const int wave = tid >> 5;
  const int half = lane >> 4;
  const int ln   = lane & 15;

  const int bh = blockIdx.y;
  const int b  = bh / H;
  const int h  = bh % H;
  const int q0 = blockIdx.x * 128;
  const int qw = q0 + wave * 16;

  __shared__ __align__(16) __bf16 lds_k[2][32][72];   // [key][d], stride 144B
  __shared__ __align__(16) __bf16 lds_v[2][64][40];   // [d][key], stride 80B
  __shared__ __bf16 lds_p[8][16][34];                 // per-wave P tile

  const int kr = tid >> 3, kc = (tid & 7) * 8;        // K tile 32x64
  const int vr = tid >> 2, vc = (tid & 3) * 8;        // Vt tile 64x32
  const __bf16* gK = Km + ((size_t)(b * T) + kr) * C + h * 64 + kc;
  const __bf16* gV = Vt + ((size_t)bh * 64 + vr) * T + vc;

  auto stage = [&](int buf, int kk0) {
    async_b128(&lds_k[buf][kr][kc], gK + (size_t)kk0 * C);
    async_b128(&lds_v[buf][vr][vc], gV + kk0);
  };

  // Q A-fragments (two d-steps)
  v16bf qa[2];
  {
    const __bf16* qrow = Q + ((size_t)(b * T + qw + ln)) * C + h * 64;
#pragma unroll
    for (int s = 0; s < 2; ++s)
#pragma unroll
      for (int v = 0; v < 8; ++v) {
        bf16x2 d = *(const bf16x2*)(qrow + s * 32 + a_k0(v, half));
        qa[s][2 * v] = d.x; qa[s][2 * v + 1] = d.y;
      }
  }

  float mrow[8], lrow[8];
#pragma unroll
  for (int r = 0; r < 8; ++r) { mrow[r] = -__builtin_inff(); lrow[r] = 0.f; }
  v8f o[4] = {{}, {}, {}, {}};

  const float scale = 0.125f;
  const int nchunk = (q0 >> 5) + 4;     // keys 0 .. q0+127

  stage(0, 0);
  for (int ic = 0; ic < nchunk; ++ic) {
    const int kk0 = ic * 32;
    const int buf = ic & 1;
    wait_async0();
    __syncthreads();
    if (ic + 1 < nchunk) stage(buf ^ 1, kk0 + 32);

    if (qw + 15 >= kk0) {               // wave-uniform: EXEC all-1 inside
      // preload all 4 K B-fragments (2 key tiles x 2 d-steps)
      v16bf bk[2][2];
#pragma unroll
      for (int c = 0; c < 2; ++c) {
        const __bf16* kp = &lds_k[buf][c * 16 + ln][0];
#pragma unroll
        for (int s = 0; s < 2; ++s)
#pragma unroll
          for (int e = 0; e < 8; ++e) {
            bf16x2 d = *(const bf16x2*)(kp + s * 32 + half * 16 + 2 * e);
            bk[c][s][2 * e] = d.x; bk[c][s][2 * e + 1] = d.y;
          }
      }
      v8f s0 = {}, s1 = {};
      s0 = WMMA_BF16(qa[0], bk[0][0], s0);
      s0 = WMMA_BF16(qa[1], bk[0][1], s0);
      s1 = WMMA_BF16(qa[0], bk[1][0], s1);
      s1 = WMMA_BF16(qa[1], bk[1][1], s1);

      // masked online softmax
      const int key0 = kk0 + ln;
      const int key1 = kk0 + 16 + ln;
#pragma unroll
      for (int r = 0; r < 8; ++r) {
        const int qg = qw + r + 8 * half;
        float a0 = s0[r] * scale; if (key0 > qg) a0 = -__builtin_inff();
        float a1 = s1[r] * scale; if (key1 > qg) a1 = -__builtin_inff();
        float mx = fmaxf(a0, a1);
        mx = fmaxf(mx, __shfl_xor(mx, 1, 32));
        mx = fmaxf(mx, __shfl_xor(mx, 2, 32));
        mx = fmaxf(mx, __shfl_xor(mx, 4, 32));
        mx = fmaxf(mx, __shfl_xor(mx, 8, 32));
        const float mnew  = fmaxf(mrow[r], mx);
        const float alpha = __expf(mrow[r] - mnew);
        const float p0 = __expf(a0 - mnew);
        const float p1 = __expf(a1 - mnew);
        float rs = p0 + p1;
        rs += __shfl_xor(rs, 1, 32);
        rs += __shfl_xor(rs, 2, 32);
        rs += __shfl_xor(rs, 4, 32);
        rs += __shfl_xor(rs, 8, 32);
        lrow[r] = lrow[r] * alpha + rs;
        mrow[r] = mnew;
        o[0][r] *= alpha; o[1][r] *= alpha; o[2][r] *= alpha; o[3][r] *= alpha;
        lds_p[wave][r + 8 * half][ln]      = f2bf(p0);
        lds_p[wave][r + 8 * half][16 + ln] = f2bf(p1);
      }

      // P as A-fragment + all 4 V B-fragments, then 4 WMMAs
      v16bf pa;
#pragma unroll
      for (int v = 0; v < 8; ++v) {
        bf16x2 d = *(const bf16x2*)&lds_p[wave][ln][a_k0(v, half)];
        pa[2 * v] = d.x; pa[2 * v + 1] = d.y;
      }
      v16bf bv[4];
#pragma unroll
      for (int c2 = 0; c2 < 4; ++c2) {
        const __bf16* cv = &lds_v[buf][c2 * 16 + ln][half * 16];
#pragma unroll
        for (int e = 0; e < 8; ++e) {
          bf16x2 d = *(const bf16x2*)(cv + 2 * e);
          bv[c2][2 * e] = d.x; bv[c2][2 * e + 1] = d.y;
        }
      }
      o[0] = WMMA_BF16(pa, bv[0], o[0]);
      o[1] = WMMA_BF16(pa, bv[1], o[1]);
      o[2] = WMMA_BF16(pa, bv[2], o[2]);
      o[3] = WMMA_BF16(pa, bv[3], o[3]);
    }
  }

  // finalize
#pragma unroll
  for (int r = 0; r < 8; ++r) {
    const float inv = lrow[r] > 0.f ? 1.f / lrow[r] : 0.f;
    const int qg = qw + r + 8 * half;
    __bf16* yr = Y + ((size_t)(b * T + qg)) * C + h * 64;
    yr[ln]      = f2bf(o[0][r] * inv);
    yr[16 + ln] = f2bf(o[1][r] * inv);
    yr[32 + ln] = f2bf(o[2][r] * inv);
    yr[48 + ln] = f2bf(o[3][r] * inv);
  }
}

// ---------------------------------------------------------------------------
extern "C" void kernel_launch(void* const* d_in, const int* in_sizes, int n_in,
                              void* d_out, int out_size, void* d_ws, size_t ws_size,
                              hipStream_t stream) {
  const int B = 4, T = 2048, C = 1024, H = 16;
  const int M = B * T;

  const float* x  = (const float*)d_in[0];
  const float* Wq = (const float*)d_in[1];
  const float* bq = (const float*)d_in[2];
  const float* Wk = (const float*)d_in[3];
  const float* bk = (const float*)d_in[4];
  const float* Wv = (const float*)d_in[5];
  const float* bv = (const float*)d_in[6];
  const float* Wo = (const float*)d_in[7];
  const float* bo = (const float*)d_in[8];
  float* out = (float*)d_out;

  char* w = (char*)d_ws;
  __bf16* Xb  = (__bf16*)w;  w += (size_t)M * C * 2;
  __bf16* Wqb = (__bf16*)w;  w += (size_t)C * C * 2;   // [N][K] transposed
  __bf16* Wkb = (__bf16*)w;  w += (size_t)C * C * 2;
  __bf16* Wvb = (__bf16*)w;  w += (size_t)C * C * 2;
  __bf16* Wob = (__bf16*)w;  w += (size_t)C * C * 2;
  __bf16* Qb  = (__bf16*)w;  w += (size_t)M * C * 2;
  __bf16* Kb  = (__bf16*)w;  w += (size_t)M * C * 2;
  __bf16* Vtb = (__bf16*)w;  w += (size_t)M * C * 2;   // [B][H][64][T]
  __bf16* Yb  = (__bf16*)w;  w += (size_t)M * C * 2;

  k_f32_to_bf16<<<2048, 256, 0, stream>>>(x, Xb, M * C);
  dim3 gt(C / 32, C / 32);
  k_convt<<<gt, 256, 0, stream>>>(Wq, Wqb, C, C);
  k_convt<<<gt, 256, 0, stream>>>(Wk, Wkb, C, C);
  k_convt<<<gt, 256, 0, stream>>>(Wv, Wvb, C, C);
  k_convt<<<gt, 256, 0, stream>>>(Wo, Wob, C, C);

  dim3 gg(C / 128, M / 128);   // (8, 64)
  k_gemm_bf16<<<gg, 256, 0, stream>>>(Xb, Wqb, bq, nullptr, Qb,  M, C, C, 1, T, H);
  k_gemm_bf16<<<gg, 256, 0, stream>>>(Xb, Wkb, bk, nullptr, Kb,  M, C, C, 1, T, H);
  k_gemm_bf16<<<gg, 256, 0, stream>>>(Xb, Wvb, bv, nullptr, Vtb, M, C, C, 2, T, H);

  dim3 ga(T / 128, B * H);
  k_attn<<<ga, 256, 0, stream>>>(Qb, Kb, Vtb, Yb, T, C, H);

  k_gemm_bf16<<<gg, 256, 0, stream>>>(Yb, Wob, bo, out, nullptr, M, C, C, 0, T, H);
}
